// ShiftFuseZero_63385127355008
// MI455X (gfx1250) — matrix operand order repaired and checked by reading, not compiled, not run on hardware
//
#include <hip/hip_runtime.h>

typedef __attribute__((ext_vector_type(16))) _Float16 v16h;
typedef __attribute__((ext_vector_type(8)))  float    v8f;

#define B_   128
#define CIN  32
#define COUT 64
#define T_   128
#define V_   25
#define K_   3

__device__ __forceinline__ float hswish(float x) {
  float t = fminf(fmaxf(x + 3.0f, 0.0f), 6.0f);
  return x * t * (1.0f / 6.0f);
}
__device__ __forceinline__ float sigmoidf(float x) {
  return 1.0f / (1.0f + __expf(-x));
}
// 16-bit A-matrix (16x32) element->K mapping for lane-half h, elem e: contiguous k = 16h + e (matches B reads)
__device__ __forceinline__ int kform(int e, int h) {
  return (h ? 16 : 0) + e;
}

// ---------------- Kernel 0: adjacency preprocessing ----------------
__global__ void k0_prep(const float* __restrict__ Af,
                        const float* __restrict__ Afix,
                        const float* __restrict__ Alrn,
                        float* __restrict__ Afn, float* __restrict__ An) {
  __shared__ float rs[V_];
  __shared__ float dkv[K_ * V_];
  int tid = threadIdx.x;
  if (tid < V_) {
    float s = 0.f;
    for (int w = 0; w < V_; w++) s += Af[tid * V_ + w];
    rs[tid] = 1.0f / fmaxf(s, 1e-6f);
  }
  if (tid < K_ * V_) {
    int k = tid / V_, v = tid % V_;
    float s = 0.f;
    for (int w = 0; w < V_; w++)
      s += fabsf(Afix[(k * V_ + v) * V_ + w] + Alrn[(k * V_ + v) * V_ + w]);
    dkv[tid] = rsqrtf(fmaxf(s, 1e-6f));
  }
  __syncthreads();
  for (int i = tid; i < V_ * V_; i += blockDim.x) {
    int v = i / V_;
    Afn[i] = Af[i] * rs[v];
  }
  for (int i = tid; i < K_ * V_ * V_; i += blockDim.x) {
    int k = i / (V_ * V_), r = i % (V_ * V_), v = r / V_, w = r % V_;
    An[i] = dkv[k * V_ + v] * fabsf(Afix[i] + Alrn[i]) * dkv[k * V_ + w];
  }
}

// ---------------- Kernel 1: BRASP + SGPShift + joint embedding ----------------
__global__ void k1_shift(const float* __restrict__ x,
                         const float* __restrict__ Aintra,
                         const float* __restrict__ Ainter,
                         const float* __restrict__ je,
                         const float* __restrict__ Afn,
                         _Float16* __restrict__ x0, float* __restrict__ Mbuf) {
  __shared__ float AfnS[V_ * V_], AitS[V_ * V_], AinS[V_ * V_];
  __shared__ float xin[CIN][V_], xa[CIN][V_];
  int t = blockIdx.x, b = blockIdx.y, tid = threadIdx.x;
  for (int i = tid; i < V_ * V_; i += blockDim.x) {
    AfnS[i] = Afn[i];
    AitS[i] = Aintra[i];
    AinS[i] = Ainter[i];
  }
  for (int i = tid; i < CIN * V_; i += blockDim.x) {
    int c = i / V_, v = i % V_;
    xin[c][v] = x[((size_t)(b * CIN + c) * T_ + t) * V_ + v];
  }
  __syncthreads();
  for (int i = tid; i < CIN * V_; i += blockDim.x) {
    int c = i / V_, w = i % V_;
    float r;
    if (c < CIN / 2) {
      r = xin[c][w];
    } else {
      r = 0.f;
      for (int v = 0; v < V_; v++) r += xin[c][v] * AfnS[v * V_ + w];
    }
    xa[c][w] = r;
  }
  __syncthreads();
  const int g = CIN / 3;  // 10
  for (int i = tid; i < CIN * V_; i += blockDim.x) {
    int c = i / V_, w = i % V_;
    float r;
    if (c < g) {
      r = xa[c][w];
    } else if (c < 2 * g) {
      r = 0.f;
      for (int v = 0; v < V_; v++) r += xa[c][v] * AitS[v * V_ + w];
    } else {
      r = 0.f;
      for (int v = 0; v < V_; v++) r += xa[c][v] * AinS[v * V_ + w];
    }
    r += je[c * V_ + w];
    xin[c][w] = r;
    x0[((size_t)(b * CIN + c) * T_ + t) * V_ + w] = (_Float16)r;
  }
  __syncthreads();
  if (tid < V_) {
    float s = 0.f;
    for (int c = 0; c < CIN; c++) s += xin[c][tid];
    Mbuf[((size_t)b * T_ + t) * V_ + tid] = s;
  }
}

// ---------------- Kernel 2a: spatial + temporal gates ----------------
__global__ void k2a_gates(const float* __restrict__ Mbuf,
                          const float* __restrict__ spw,
                          const float* __restrict__ tw,
                          float* __restrict__ g1, float* __restrict__ g2) {
  __shared__ float Ms[T_][V_];
  __shared__ float sA[V_], lA[V_], eA[V_], red[2];
  __shared__ float tm[T_];
  int b = blockIdx.x, tid = threadIdx.x;  // blockDim = 128
  for (int i = tid; i < T_ * V_; i += blockDim.x)
    Ms[i / V_][i % V_] = Mbuf[(size_t)b * T_ * V_ + i];
  __syncthreads();
  if (tid < V_) {
    float s = 0.f;
    for (int t = 0; t < T_; t++) s += Ms[t][tid];
    sA[tid] = s * (1.0f / (CIN * T_));
  }
  __syncthreads();
  if (tid < V_) {
    float s = 0.f;
    for (int v = 0; v < V_; v++) s += sA[v] * spw[v * V_ + tid];
    lA[tid] = s;
  }
  __syncthreads();
  if (tid == 0) {
    float m = lA[0];
    for (int w = 1; w < V_; w++) m = fmaxf(m, lA[w]);
    red[0] = m;
  }
  __syncthreads();
  if (tid < V_) eA[tid] = __expf(lA[tid] - red[0]);
  __syncthreads();
  if (tid == 0) {
    float s = 0.f;
    for (int w = 0; w < V_; w++) s += eA[w];
    red[1] = s;
  }
  __syncthreads();
  if (tid < V_) {
    float gv = 1.0f + eA[tid] / red[1];
    sA[tid] = gv;
    g1[b * V_ + tid] = gv;
  }
  __syncthreads();
  if (tid < T_) {
    float s = 0.f;
    for (int v = 0; v < V_; v++) s += Ms[tid][v] * sA[v];
    tm[tid] = s * (1.0f / (CIN * V_));
  }
  __syncthreads();
  if (tid < T_) {
    float s = 0.f;
    for (int i = 0; i < 9; i++) {
      int tt = tid + i - 4;
      if (tt >= 0 && tt < T_) s += tm[tt] * tw[i];
    }
    g2[b * T_ + tid] = sigmoidf(s);
  }
}

// ---------------- Kernel 2b: channel SE gate ----------------
__global__ void k2b_se(const _Float16* __restrict__ x0,
                       const float* __restrict__ g1,
                       const float* __restrict__ g2,
                       const float* __restrict__ w1,
                       const float* __restrict__ w2, float* __restrict__ g3) {
  __shared__ float g1s[V_], g2s[T_], part[256], cmS[CIN], h1S[CIN / 4];
  int b = blockIdx.x, tid = threadIdx.x;  // blockDim = 256
  if (tid < V_) g1s[tid] = g1[b * V_ + tid];
  if (tid < T_) g2s[tid] = g2[b * T_ + tid];
  __syncthreads();
  int c = tid >> 3, s8 = tid & 7;
  const _Float16* xp = x0 + (size_t)(b * CIN + c) * T_ * V_;
  float acc = 0.f;
  for (int i = s8; i < T_ * V_; i += 8) {
    int t = i / V_, v = i % V_;
    acc += (float)xp[i] * g1s[v] * g2s[t];
  }
  part[tid] = acc;
  __syncthreads();
  if (tid < CIN) {
    float s = 0.f;
    for (int j = 0; j < 8; j++) s += part[tid * 8 + j];
    cmS[tid] = s * (1.0f / (T_ * V_));
  }
  __syncthreads();
  if (tid < CIN / 4) {
    float s = 0.f;
    for (int cc = 0; cc < CIN; cc++) s += cmS[cc] * w1[cc * (CIN / 4) + tid];
    h1S[tid] = hswish(s);
  }
  __syncthreads();
  if (tid < CIN) {
    float s = 0.f;
    for (int j = 0; j < CIN / 4; j++) s += h1S[j] * w2[j * CIN + tid];
    g3[b * CIN + tid] = sigmoidf(s);
  }
}

// ---------------- Kernel 3: gated GCN, both contractions on WMMA ----------------
// block = (16-t chunk, batch). u_k = x' A_k via WMMA (M=(c,t)=512,K=v pad32,N=w pad32),
// then y = hswish(BN((1/K) sum_k W_k @ u_k)) via WMMA (M=64,N=400,K=32).
__global__ void __launch_bounds__(256) k3_gcn(
    const _Float16* __restrict__ x0, const float* __restrict__ An,
    const float* __restrict__ Wg, const float* __restrict__ g1,
    const float* __restrict__ g2, const float* __restrict__ g3,
    const float* __restrict__ gs, const float* __restrict__ gb,
    _Float16* __restrict__ y) {
  __shared__ _Float16 xsA[512 * 32];     // pre-shuffled A-layout: [row][h][e]
  __shared__ _Float16 uT[400][32];       // [n=(tl*25+w)][c]  -- K-innermost
  __shared__ _Float16 WkA[COUT * 2 * 16];// pre-shuffled A-layout: [row][h][e]
  __shared__ _Float16 AnT[32][32];       // [w][v], zero padded
  __shared__ float g1s[V_], g2l[16], g3s[CIN], gsS[COUT], gbS[COUT];
  int chunk = blockIdx.x, b = blockIdx.y, tid = threadIdx.x;
  int t0 = chunk * 16;
  if (tid < V_) g1s[tid] = g1[b * V_ + tid];
  if (tid < 16) g2l[tid] = g2[b * T_ + t0 + tid];
  if (tid < CIN) g3s[tid] = g3[b * CIN + tid];
  if (tid < COUT) {
    gsS[tid] = gs[tid];
    gbS[tid] = gb[tid];
  }
  __syncthreads();
  // gated x', pre-shuffled into the A-fragment element order (zero-padded v>=25).
  // One wave builds one row; source is a single 50B x0 row -> coalesced.
  for (int i = tid; i < 512 * 32; i += 256) {
    int row = i >> 5, j = i & 31, hh = j >> 4, e = j & 15;
    int c = row >> 4, tl = row & 15;
    int v = kform(e, hh);
    float val = 0.f;
    if (v < V_) {
      val = (float)x0[((size_t)(b * CIN + c) * T_ + t0 + tl) * V_ + v] *
            g1s[v] * g2l[tl] * g3s[c];
    }
    xsA[i] = (_Float16)val;
  }

  int wid = tid >> 5, lane = tid & 31;
  int mt = wid & 3, ph = wid >> 2;
  bool hi = lane >= 16;
  int h = hi ? 1 : 0;
  int mrow = lane & 15;

  v8f zero8 = {0.f, 0.f, 0.f, 0.f, 0.f, 0.f, 0.f, 0.f};
  v8f acc[13];
#pragma unroll
  for (int j = 0; j < 13; j++) acc[j] = zero8;

  for (int k = 0; k < K_; k++) {
    __syncthreads();  // xsA ready / previous GEMM done with uT & WkA
    // stage A_k (B-operand, [w][v] so K=v is contiguous) and shuffled W_k
    for (int i = tid; i < 1024; i += 256) {
      int w = i >> 5, v = i & 31;
      float a = (v < V_ && w < V_) ? An[k * V_ * V_ + v * V_ + w] : 0.f;
      AnT[w][v] = (_Float16)a;
    }
    for (int i = tid; i < COUT * 32; i += 256) {
      int row = i >> 5, j5 = i & 31, hh = j5 >> 4, e = j5 & 15;
      WkA[i] = (_Float16)Wg[k * COUT * CIN + row * CIN + kform(e, hh)];
    }
    __syncthreads();
    // --- graph GEMM: u_k[(c,tl), w] ---  (each M-tile == one channel)
    for (int mi = 0; mi < 4; mi++) {
      int c = wid * 4 + mi;
      int row = c * 16 + mrow;
      const v16h a = *(const v16h*)&xsA[(row * 2 + h) * 16];
#pragma unroll
      for (int ntile = 0; ntile < 2; ntile++) {
        const v16h bb = *(const v16h*)&AnT[ntile * 16 + mrow][h * 16];
        v8f du = zero8;
        du = __builtin_amdgcn_wmma_f32_16x16x32_f16(false, a, false, bb,
                                                    (short)0, du, false, false);
        int nw = ntile * 16 + mrow;
        if (nw < V_) {  // per-lane store guard (plain ds stores)
#pragma unroll
          for (int r = 0; r < 8; r++) {
            int tl = r + (hi ? 8 : 0);
            uT[tl * V_ + nw][c] = (_Float16)du[r];
          }
        }
      }
    }
    __syncthreads();
    // --- channel GEMM: acc += W_k @ u_k ---
    const v16h aw = *(const v16h*)&WkA[((mt * 16 + mrow) * 2 + h) * 16];
#pragma unroll
    for (int j = 0; j < 13; j++) {
      int nt = ph + 2 * j;
      if (nt < 25) {  // wave-uniform
        int n = nt * 16 + mrow;
        const v16h bb = *(const v16h*)&uT[n][h * 16];
        acc[j] = __builtin_amdgcn_wmma_f32_16x16x32_f16(
            false, aw, false, bb, (short)0, acc[j], false, false);
      }
    }
  }
  // epilogue: /K, BN, hardswish -> y (f16)
#pragma unroll
  for (int j = 0; j < 13; j++) {
    int nt = ph + 2 * j;
    if (nt < 25) {
      int n = nt * 16 + mrow;
      int tl = n / V_, w = n % V_;
#pragma unroll
      for (int r = 0; r < 8; r++) {
        int o = mt * 16 + r + (hi ? 8 : 0);
        float val = acc[j][r] * (1.0f / K_) * gsS[o] + gbS[o];
        y[((size_t)(b * COUT + o) * T_ + t0 + tl) * V_ + w] =
            (_Float16)hswish(val);
      }
    }
  }
}

// ---------------- Kernel 4: depthwise TCN + pointwise + residual via WMMA ----------------
__global__ void __launch_bounds__(256) k4_tcn(
    const float* __restrict__ x, const _Float16* __restrict__ y,
    const float* __restrict__ dww, const float* __restrict__ pw,
    const float* __restrict__ resw, const float* __restrict__ ts,
    const float* __restrict__ tb, const float* __restrict__ rs,
    const float* __restrict__ rb, float* __restrict__ out) {
  __shared__ _Float16 dwT[208][COUT];      // [n][c] K-innermost
  __shared__ _Float16 xinT[208][CIN];      // [n][c]
  __shared__ _Float16 pwA[COUT * 2 * 2 * 16];  // [row][s][h][e], A-shuffled
  __shared__ _Float16 resA[COUT * 2 * 16];     // [row][h][e]
  __shared__ float tsS[COUT], tbS[COUT], rsS[COUT], rbS[COUT];
  int chunk = blockIdx.x, b = blockIdx.y, tid = threadIdx.x;
  int t0p = chunk * 8;
  if (tid < COUT) {
    tsS[tid] = ts[tid];
    tbS[tid] = tb[tid];
    rsS[tid] = rs[tid];
    rbS[tid] = rb[tid];
  }
  for (int i = tid; i < COUT * 64; i += 256) {
    int row = i >> 6, j6 = i & 63, s = j6 >> 5, j5 = j6 & 31, hh = j5 >> 4,
        e = j5 & 15;
    pwA[i] = (_Float16)pw[row * COUT + s * 32 + kform(e, hh)];
  }
  for (int i = tid; i < COUT * 32; i += 256) {
    int row = i >> 5, j5 = i & 31, hh = j5 >> 4, e = j5 & 15;
    resA[i] = (_Float16)resw[row * CIN + kform(e, hh)];
  }
  // depthwise 5-tap stride-2 conv (coalesced y reads, transposed LDS writes)
  for (int i = tid; i < COUT * 208; i += 256) {
    int c = i / 208, n = i - c * 208;
    float r = 0.f;
    if (n < 200) {
      int tl = n / V_, w = n % V_, tp = t0p + tl;
      for (int q = 0; q < 5; q++) {
        int tt = 2 * tp + q - 2;
        if (tt >= 0 && tt < T_)
          r += (float)y[((size_t)(b * COUT + c) * T_ + tt) * V_ + w] *
               dww[c * 5 + q];
      }
    }
    dwT[n][c] = (_Float16)r;
  }
  for (int i = tid; i < CIN * 208; i += 256) {
    int c = i / 208, n = i - c * 208;
    float r = 0.f;
    if (n < 200) {
      int tl = n / V_, w = n % V_;
      r = x[((size_t)(b * CIN + c) * T_ + 2 * (t0p + tl)) * V_ + w];
    }
    xinT[n][c] = (_Float16)r;
  }
  __syncthreads();

  int wid = tid >> 5, lane = tid & 31;
  int mt = wid & 3, ph = wid >> 2;
  bool hi = lane >= 16;
  int h = hi ? 1 : 0;
  int mrow = lane & 15;
  v8f zero8 = {0.f, 0.f, 0.f, 0.f, 0.f, 0.f, 0.f, 0.f};

  int row = mt * 16 + mrow;
  const v16h apw0 = *(const v16h*)&pwA[((row * 2 + 0) * 2 + h) * 16];
  const v16h apw1 = *(const v16h*)&pwA[((row * 2 + 1) * 2 + h) * 16];
  const v16h ares = *(const v16h*)&resA[(row * 2 + h) * 16];

  for (int j = 0; j < 7; j++) {
    int nt = ph + 2 * j;
    if (nt >= 13) break;  // wave-uniform
    int n = nt * 16 + mrow;
    const v16h b0 = *(const v16h*)&dwT[n][h * 16];
    const v16h b1 = *(const v16h*)&dwT[n][32 + h * 16];
    const v16h bx = *(const v16h*)&xinT[n][h * 16];
    v8f aM = zero8, aR = zero8;
    aM = __builtin_amdgcn_wmma_f32_16x16x32_f16(false, apw0, false, b0,
                                                (short)0, aM, false, false);
    aM = __builtin_amdgcn_wmma_f32_16x16x32_f16(false, apw1, false, b1,
                                                (short)0, aM, false, false);
    aR = __builtin_amdgcn_wmma_f32_16x16x32_f16(false, ares, false, bx,
                                                (short)0, aR, false, false);
    if (n < 200) {  // per-lane store guard (after WMMA)
      int tl = n / V_, w = n % V_, tp = t0p + tl;
#pragma unroll
      for (int r = 0; r < 8; r++) {
        int o = mt * 16 + r + (hi ? 8 : 0);
        float zM = hswish(aM[r] * tsS[o] + tbS[o]);
        float zR = aR[r] * rsS[o] + rbS[o];
        out[((size_t)(b * COUT + o) * (T_ / 2) + tp) * V_ + w] = hswish(zM + zR);
      }
    }
  }
}

extern "C" void kernel_launch(void* const* d_in, const int* in_sizes, int n_in,
                              void* d_out, int out_size, void* d_ws,
                              size_t ws_size, hipStream_t stream) {
  (void)in_sizes; (void)n_in; (void)out_size; (void)ws_size;
  const float* x      = (const float*)d_in[0];
  const float* Aflat  = (const float*)d_in[1];
  const float* Aintra = (const float*)d_in[2];
  const float* Ainter = (const float*)d_in[3];
  const float* Afix   = (const float*)d_in[4];
  const float* Alrn   = (const float*)d_in[5];
  const float* Wg     = (const float*)d_in[6];
  const float* gs     = (const float*)d_in[7];
  const float* gb     = (const float*)d_in[8];
  const float* je     = (const float*)d_in[9];
  const float* spw    = (const float*)d_in[10];
  const float* tw     = (const float*)d_in[11];
  const float* sew1   = (const float*)d_in[12];
  const float* sew2   = (const float*)d_in[13];
  const float* dww    = (const float*)d_in[14];
  const float* pww    = (const float*)d_in[15];
  const float* tbs    = (const float*)d_in[16];
  const float* tbb    = (const float*)d_in[17];
  const float* resw   = (const float*)d_in[18];
  const float* rbs    = (const float*)d_in[19];
  const float* rbb    = (const float*)d_in[20];
  float* out = (float*)d_out;

  char* ws = (char*)d_ws;
  size_t p = 0;
  auto carve = [&](size_t bytes) -> char* {
    char* r = ws + p;
    p = (p + bytes + 255) & ~(size_t)255;
    return r;
  };
  float* Afn  = (float*)carve(V_ * V_ * 4);
  float* An   = (float*)carve(K_ * V_ * V_ * 4);
  float* Mbuf = (float*)carve((size_t)B_ * T_ * V_ * 4);
  float* g1   = (float*)carve((size_t)B_ * V_ * 4);
  float* g2   = (float*)carve((size_t)B_ * T_ * 4);
  float* g3   = (float*)carve((size_t)B_ * CIN * 4);
  _Float16* x0 = (_Float16*)carve((size_t)B_ * CIN * T_ * V_ * 2);
  _Float16* yb = (_Float16*)carve((size_t)B_ * COUT * T_ * V_ * 2);

  k0_prep<<<1, 256, 0, stream>>>(Aflat, Afix, Alrn, Afn, An);
  k1_shift<<<dim3(T_, B_), 256, 0, stream>>>(x, Aintra, Ainter, je, Afn, x0, Mbuf);
  k2a_gates<<<B_, 128, 0, stream>>>(Mbuf, spw, tw, g1, g2);
  k2b_se<<<B_, 256, 0, stream>>>(x0, g1, g2, sew1, sew2, g3);
  k3_gcn<<<dim3(8, B_), 256, 0, stream>>>(x0, An, Wg, g1, g2, g3, gs, gb, yb);
  k4_tcn<<<dim3(8, B_), 256, 0, stream>>>(x, yb, dww, pww, resw, tbs, tbb, rbs, rbb, out);
}
